// Encoder_27015344292106
// MI455X (gfx1250) — matrix operand order repaired and checked by reading, not compiled, not run on hardware
//
#include <hip/hip_runtime.h>
#include <math.h>

// ---------- CDNA5 WMMA types ----------
typedef __attribute__((ext_vector_type(16))) __bf16 v16bf;
typedef __attribute__((ext_vector_type(8)))  float  v8f;

union BF16x4 { uint2 u; __bf16 h[4]; };
union FragBF { v16bf v; uint2 u2[4]; uint4 u4[2]; };

#define BM 128
#define BN 128
#define BK 32
#define LDT 40   // padded LDS row pitch in bf16 elements (80B, 16B aligned)

// Async global->LDS 16-byte copy (gfx1250, ASYNCcnt-tracked, no VGPR data path).
// Low 32 bits of a generic shared pointer are the LDS byte offset (ISA 10.2 aperture map).
__device__ __forceinline__ void async_b128(const void* g, const void* l) {
    unsigned loff = (unsigned)(uintptr_t)l;
    asm volatile("global_load_async_to_lds_b128 %0, %1, off"
                 :: "v"(loff), "v"(g) : "memory");
}
__device__ __forceinline__ void wait_async0() {
    asm volatile("s_wait_asynccnt 0x0" ::: "memory");
}

// Generic tiled GEMM: C[M,N] = act( scale * (A[M,K] @ B) + bias ), WMMA bf16 -> f32.
// Block: 256 threads = 8 waves, BMxBN = 128x128, wave tile 32x64 (8 WMMA / K-step).
//  A_BF16 : A is bf16 row-major [M,K] (async-to-LDS path), else fp32 (convert while staging)
//  B_BF16 : B element type
//  B_TRANS: B stored as [N,K] row-major (compute A @ B^T); bf16 -> async-to-LDS path
//  RELU   : apply max(0,x)
//  OUT    : 0 = f32 only, 1 = bf16 only, 2 = both
template<bool A_BF16, bool B_BF16, bool B_TRANS, bool RELU, int OUT>
__global__ __launch_bounds__(256)
void gemm_wmma(const void* __restrict__ Ap, const void* __restrict__ Bp,
               const float* __restrict__ bias, float scale,
               float* __restrict__ Cf, __bf16* __restrict__ Cb,
               int M, int N, int K,
               long long strA, long long strB, long long strC)
{
    constexpr bool ASYNC_A = A_BF16;
    constexpr bool ASYNC_B = (B_TRANS && B_BF16);

    __shared__ __align__(16) __bf16 tA[2][BM * LDT];
    __shared__ __align__(16) __bf16 tB[2][BN * LDT];

    const int tid  = threadIdx.x;
    const int lane = tid & 31;
    const int wave = tid >> 5;          // 0..7
    const int wm   = wave >> 1;         // 0..3 : rows of 32
    const int wn   = wave & 1;          // 0..1 : cols of 64
    const int bm0  = blockIdx.y * BM;
    const int bn0  = blockIdx.x * BN;
    const int bat  = blockIdx.z;

    const float*  Af = (const float* )Ap + (size_t)bat * strA;
    const __bf16* Ab = (const __bf16*)Ap + (size_t)bat * strA;
    const float*  Bf = (const float* )Bp + (size_t)bat * strB;
    const __bf16* Bb = (const __bf16*)Bp + (size_t)bat * strB;
    if (OUT == 0 || OUT == 2) Cf += (size_t)bat * strC;
    if (OUT == 1 || OUT == 2) Cb += (size_t)bat * strC;

    // ---- per-thread staging pointers (advanced by constant per K-step) ----
    const float*  pAf[4];  int sA_lds[4];                 // A fp32: 4 x float4
    const __bf16* pAb[2];  const __bf16* lAa[2][2];       // A bf16 async: 2 x 16B
    const float*  pBf[4];  const __bf16* pBb[4];          // B [K,N]: 4 x 4-elem groups
    int           krB[4], c4B[4];
    const __bf16* pBa[2];  const __bf16* lBa[2][2];       // B [N,K] bf16 async: 2 x 16B

#pragma unroll
    for (int i = 0; i < 4; ++i) {
        int idx = tid + i * 256;
        { int r = idx >> 3, c4 = (idx & 7) << 2;          // A: 128 rows x 8 groups
          pAf[i]    = Af + (size_t)(bm0 + r) * K + c4;
          sA_lds[i] = r * LDT + c4; }
        { int kr = idx >> 5, c4 = (idx & 31) << 2;        // B: 32 rows x 32 groups
          pBf[i] = Bf + (size_t)kr * N + bn0 + c4;
          pBb[i] = Bb + (size_t)kr * N + bn0 + c4;
          krB[i] = kr; c4B[i] = c4; }
    }
#pragma unroll
    for (int i = 0; i < 2; ++i) {
        int idx = tid + i * 256;
        int r = idx >> 2, c8 = (idx & 3) << 3;            // 128 rows x 4 chunks
        pAb[i] = Ab + (size_t)(bm0 + r) * K + c8;
        lAa[0][i] = &tA[0][r * LDT + c8];
        lAa[1][i] = &tA[1][r * LDT + c8];
        pBa[i] = Bb + (size_t)(bn0 + r) * K + c8;
        lBa[0][i] = &tB[0][r * LDT + c8];
        lBa[1][i] = &tB[1][r * LDT + c8];
    }

    float4 arf[4];                  // staged fp32 A rows
    float4 brf[4]; BF16x4 brb[4];   // staged B groups

    auto issue = [&](int buf) {
        if (ASYNC_A) {
#pragma unroll
            for (int i = 0; i < 2; ++i) { async_b128(pAb[i], lAa[buf][i]); pAb[i] += BK; }
        } else {
#pragma unroll
            for (int i = 0; i < 4; ++i) { arf[i] = *(const float4*)pAf[i]; pAf[i] += BK; }
        }
        if (ASYNC_B) {
#pragma unroll
            for (int i = 0; i < 2; ++i) { async_b128(pBa[i], lBa[buf][i]); pBa[i] += BK; }
        } else {
#pragma unroll
            for (int i = 0; i < 4; ++i) {
                if (B_BF16) { brb[i].u = *(const uint2*)pBb[i]; pBb[i] += (size_t)BK * N; }
                else        { brf[i]   = *(const float4*)pBf[i]; pBf[i] += (size_t)BK * N; }
            }
        }
    };
    auto commit = [&](int buf) {
        if (!ASYNC_A) {
#pragma unroll
            for (int i = 0; i < 4; ++i) {
                BF16x4 w;
                w.h[0] = (__bf16)arf[i].x; w.h[1] = (__bf16)arf[i].y;
                w.h[2] = (__bf16)arf[i].z; w.h[3] = (__bf16)arf[i].w;
                *(uint2*)&tA[buf][sA_lds[i]] = w.u;
            }
        }
        if (!ASYNC_B) {
#pragma unroll
            for (int i = 0; i < 4; ++i) {
                __bf16 h0, h1, h2, h3;
                if (B_BF16) { h0 = brb[i].h[0]; h1 = brb[i].h[1]; h2 = brb[i].h[2]; h3 = brb[i].h[3]; }
                else { h0 = (__bf16)brf[i].x; h1 = (__bf16)brf[i].y;
                       h2 = (__bf16)brf[i].z; h3 = (__bf16)brf[i].w; }
                tB[buf][(c4B[i] + 0) * LDT + krB[i]] = h0;
                tB[buf][(c4B[i] + 1) * LDT + krB[i]] = h1;
                tB[buf][(c4B[i] + 2) * LDT + krB[i]] = h2;
                tB[buf][(c4B[i] + 3) * LDT + krB[i]] = h3;
            }
        }
        if (ASYNC_A || ASYNC_B) wait_async0();
    };

    v8f acc[2][4];
#pragma unroll
    for (int i = 0; i < 2; ++i)
#pragma unroll
        for (int j = 0; j < 4; ++j) acc[i][j] = v8f{0.f,0.f,0.f,0.f,0.f,0.f,0.f,0.f};

    const int r16   = lane & 15;
    const int khalf = lane >> 4;
    const int nK    = K / BK;

    // ---- software pipeline ----
    issue(0);
    commit(0);
    __syncthreads();

    for (int kt = 0; kt < nK; ++kt) {
        const int c = kt & 1;
        if (kt + 1 < nK) issue(c ^ 1);   // async writes go straight to the other buffer

        // ---- fragments (ISA 7.12.2 layouts) from buffer c ----
        FragBF a[2], b[4];
#pragma unroll
        for (int mi = 0; mi < 2; ++mi) {
            const __bf16* ap = &tA[c][(wm * 32 + mi * 16 + r16) * LDT + khalf * 8];
            a[mi].u2[0] = *(const uint2*)(ap);      a[mi].u2[1] = *(const uint2*)(ap + 4);
            a[mi].u2[2] = *(const uint2*)(ap + 16); a[mi].u2[3] = *(const uint2*)(ap + 20);
        }
#pragma unroll
        for (int ni = 0; ni < 4; ++ni) {
            const __bf16* bp = &tB[c][(wn * 64 + ni * 16 + r16) * LDT + khalf * 8];
            b[ni].u4[0] = *(const uint4*)(bp); b[ni].u4[1] = *(const uint4*)(bp + 16);
        }

#pragma unroll
        for (int mi = 0; mi < 2; ++mi)
#pragma unroll
            for (int ni = 0; ni < 4; ++ni)
                acc[mi][ni] = __builtin_amdgcn_wmma_f32_16x16x32_bf16(
                    false, a[mi].v, false, b[ni].v, (short)0, acc[mi][ni], false, false);

        if (kt + 1 < nK) {
            commit(c ^ 1);
            __syncthreads();
        }
    }

    // ---- epilogue: C/D layout -> VGPR r holds M = r + 8*(lane>=16), N = lane&15 ----
    const int nl = lane & 15;
#pragma unroll
    for (int mi = 0; mi < 2; ++mi) {
#pragma unroll
        for (int ni = 0; ni < 4; ++ni) {
            v8f cacc = acc[mi][ni];
            int col  = bn0 + wn * 64 + ni * 16 + nl;
            int row0 = bm0 + wm * 32 + mi * 16 + khalf * 8;
            float bv = bias ? bias[col] : 0.f;
#pragma unroll
            for (int r = 0; r < 8; ++r) {
                float v = cacc[r] * scale + bv;
                if (RELU) v = fmaxf(v, 0.f);
                size_t off = (size_t)(row0 + r) * N + col;
                if (OUT == 0 || OUT == 2) Cf[off] = v;
                if (OUT == 1 || OUT == 2) Cb[off] = (__bf16)v;
            }
        }
    }
    (void)M;
}

// ---------- row softmax: fp32 scores -> bf16 attention ----------
__global__ __launch_bounds__(256)
void softmax_rows(const float* __restrict__ S, __bf16* __restrict__ A, int n)
{
    __shared__ float red[256];
    const int tid = threadIdx.x;
    const float* p = S + (size_t)blockIdx.x * n;
    __bf16*      q = A + (size_t)blockIdx.x * n;

    float m = -3.402823466e+38f;
    for (int j = tid; j < n; j += 256) m = fmaxf(m, p[j]);
    red[tid] = m; __syncthreads();
    for (int s = 128; s > 0; s >>= 1) { if (tid < s) red[tid] = fmaxf(red[tid], red[tid + s]); __syncthreads(); }
    m = red[0]; __syncthreads();

    float sum = 0.f;
    for (int j = tid; j < n; j += 256) sum += __expf(p[j] - m);
    red[tid] = sum; __syncthreads();
    for (int s = 128; s > 0; s >>= 1) { if (tid < s) red[tid] += red[tid + s]; __syncthreads(); }
    float inv = 1.f / red[0];

    for (int j = tid; j < n; j += 256) q[j] = (__bf16)(__expf(p[j] - m) * inv);
}

// ---------- fused residual-add + LayerNorm (eps = 1e-3) ----------
template<bool WRITE_BF16>
__global__ __launch_bounds__(256)
void add_layernorm(const float* __restrict__ A, const float* __restrict__ Bv,
                   const float* __restrict__ gamma, const float* __restrict__ beta,
                   float* __restrict__ outF, __bf16* __restrict__ outB, int n)
{
    __shared__ float red[256];
    __shared__ float z[1024];
    const int tid = threadIdx.x;
    const size_t base = (size_t)blockIdx.x * n;

    float s = 0.f;
    for (int j = tid; j < n; j += 256) {
        float v = A[base + j] + Bv[base + j];
        z[j] = v; s += v;
    }
    red[tid] = s; __syncthreads();
    for (int t = 128; t > 0; t >>= 1) { if (tid < t) red[tid] += red[tid + t]; __syncthreads(); }
    float mu = red[0] * (1.f / n); __syncthreads();

    float vs = 0.f;
    for (int j = tid; j < n; j += 256) { float d = z[j] - mu; vs += d * d; }
    red[tid] = vs; __syncthreads();
    for (int t = 128; t > 0; t >>= 1) { if (tid < t) red[tid] += red[tid + t]; __syncthreads(); }
    float rstd = rsqrtf(red[0] * (1.f / n) + 1e-3f);

    for (int j = tid; j < n; j += 256) {
        float o = gamma[j] * (z[j] - mu) * rstd + beta[j];
        outF[base + j] = o;
        if (WRITE_BF16) outB[base + j] = (__bf16)o;
    }
}

extern "C" void kernel_launch(void* const* d_in, const int* in_sizes, int n_in,
                              void* d_out, int out_size, void* d_ws, size_t ws_size,
                              hipStream_t stream)
{
    const float* x     = (const float*)d_in[0];
    const float* Wq    = (const float*)d_in[1];
    const float* bq    = (const float*)d_in[2];
    const float* Wk    = (const float*)d_in[3];
    const float* bk    = (const float*)d_in[4];
    // d_in[5]=Wv, d_in[6]=bv : unused by reference (value path uses K)
    const float* Wo    = (const float*)d_in[7];
    const float* bo    = (const float*)d_in[8];
    const float* W1    = (const float*)d_in[9];
    const float* b1    = (const float*)d_in[10];
    const float* W2    = (const float*)d_in[11];
    const float* b2    = (const float*)d_in[12];
    const float* gamma = (const float*)d_in[13];
    const float* beta  = (const float*)d_in[14];

    const int Bn = 8, S = 1024, E = 1024;
    const int M  = Bn * S;            // 8192
    const int F  = 4 * E;             // 4096
    const float scl = 0.03125f;       // 1/sqrt(1024)

    char* ws = (char*)d_ws;
    const size_t MB = 1024ull * 1024ull;
    __bf16* Qb  = (__bf16*)(ws + 0);        //  16 MB  Q bf16
    __bf16* Kb  = (__bf16*)(ws + 16 * MB);  //  16 MB  K bf16
    float*  Sf  = (float* )(ws + 32 * MB);  //  32 MB  scores -> y -> f (reused)
    __bf16* At  = (__bf16*)(ws + 64 * MB);  //  16 MB  attn bf16
    __bf16* Cx  = (__bf16*)(ws + 80 * MB);  //  16 MB  ctx bf16
    float*  R2f = (float* )(ws + 96 * MB);  //  32 MB  res2 f32
    __bf16* R2b = (__bf16*)(ws + 128 * MB); //  16 MB  res2 bf16
    __bf16* Hb  = (__bf16*)(ws + 144 * MB); //  64 MB  h bf16   (peak 208 MB)

    dim3 blk(256);
    const long long sSE = (long long)S * E;
    const long long sSS = (long long)S * S;

    // 1-2) Q = x@Wq+bq, K = x@Wk+bk  (fp32 in, bf16 out)
    gemm_wmma<false,false,false,false,1><<<dim3(E/BN, M/BM, 1), blk, 0, stream>>>(
        x, Wq, bq, 1.f, nullptr, Qb, M, E, E, 0, 0, 0);
    gemm_wmma<false,false,false,false,1><<<dim3(E/BN, M/BM, 1), blk, 0, stream>>>(
        x, Wk, bk, 1.f, nullptr, Kb, M, E, E, 0, 0, 0);

    // 3) scores = Q @ K^T / 32  (batched NT, both operands async-to-LDS, f32 out)
    gemm_wmma<true,true,true,false,0><<<dim3(S/BN, S/BM, Bn), blk, 0, stream>>>(
        Qb, Kb, nullptr, scl, Sf, nullptr, S, S, E, sSE, sSE, sSS);

    // 4) attn = softmax(scores)  (bf16 out)
    softmax_rows<<<dim3(M), dim3(256), 0, stream>>>(Sf, At, S);

    // 5) ctx = attn @ K  (batched NN, A async, bf16 out)
    gemm_wmma<true,true,false,false,1><<<dim3(E/BN, S/BM, Bn), blk, 0, stream>>>(
        At, Kb, nullptr, 1.f, nullptr, Cx, S, E, S, sSS, sSE, sSE);

    // 6) y = ctx @ Wo + bo  (A async, f32 out, reuses scores region)
    gemm_wmma<true,false,false,false,0><<<dim3(E/BN, M/BM, 1), blk, 0, stream>>>(
        Cx, Wo, bo, 1.f, Sf, nullptr, M, E, E, 0, 0, 0);

    // 7) res2 = LN(y + x)  (f32 + bf16 out)
    add_layernorm<true><<<dim3(M), dim3(256), 0, stream>>>(
        Sf, x, gamma, beta, R2f, R2b, E);

    // 8) h = relu(res2 @ W1 + b1)  (A async, bf16 out)
    gemm_wmma<true,false,false,true,1><<<dim3(F/BN, M/BM, 1), blk, 0, stream>>>(
        R2b, W1, b1, 1.f, nullptr, Hb, M, F, E, 0, 0, 0);

    // 9) f = h @ W2 + b2  (A async, f32 out, reuses scores region)
    gemm_wmma<true,false,false,false,0><<<dim3(E/BN, M/BM, 1), blk, 0, stream>>>(
        Hb, W2, b2, 1.f, Sf, nullptr, M, E, F, 0, 0, 0);

    // 10) out = LN(f + res2)
    add_layernorm<false><<<dim3(M), dim3(256), 0, stream>>>(
        Sf, R2f, gamma, beta, (float*)d_out, nullptr, E);

    (void)in_sizes; (void)n_in; (void)out_size; (void)ws_size;
}
